// ModernBERT_4432406249955
// MI455X (gfx1250) — matrix operand order repaired and checked by reading, not compiled, not run on hardware
//
#include <hip/hip_runtime.h>
#include <cmath>

// ---------------------------------------------------------------------------
// ModernBERT forward on gfx1250 (MI455X), wave32 + WMMA bf16.
// All matmuls run through v_wmma_f32_16x16x32_bf16 with fp32 accumulation.
// fp32 operands are converted to bf16 while staging into LDS (b128 global
// loads, packed b32 LDS stores). LDS is double-buffered: the global loads for
// K-step s+1 issue before the WMMAs of step s, one barrier per step.
// Leading dimensions are template parameters -> 32-bit shift/add addressing.
// ---------------------------------------------------------------------------

typedef __attribute__((ext_vector_type(16))) __bf16        v16bf;
typedef __attribute__((ext_vector_type(8)))  float         v8f;
typedef __attribute__((ext_vector_type(8)))  unsigned int  v8u;

#define HID   768
#define NL    22
#define NH    12
#define DH    64
#define FF    1152
#define SEQ   512
#define BATCH 4
#define TOK   (BATCH * SEQ)   // 2048
#define NEGB  (-1e9f)

constexpr int FLAG_BTRANS = 1;
constexpr int FLAG_GEGLU  = 2;
constexpr int FLAG_RES    = 4;

__device__ __forceinline__ unsigned pack_bf16(float x, float y) {
    unsigned short lo = __builtin_bit_cast(unsigned short, (__bf16)x);
    unsigned short hi = __builtin_bit_cast(unsigned short, (__bf16)y);
    return (unsigned)lo | ((unsigned)hi << 16);
}

__device__ __forceinline__ float gelu_exact(float x) {
    return 0.5f * x * (1.0f + erff(x * 0.70710678f));
}

// ---------------------------------------------------------------------------
// Generic WMMA GEMM:  C[M,N] = A[M,K] * B[K,N]  (FLAG_BTRANS: B given [N,K]),
// A optionally produced as gelu(A)*A2 (FLAG_GEGLU), epilogue +Res (FLAG_RES).
// Batched over blockIdx.z = b*nh + h with independent strides.
// 256 threads = 8 waves; wave computes WM x WN via 16x16x32 bf16 WMMA tiles.
// ---------------------------------------------------------------------------
template<int BM, int BN, int WM, int WN, int LDA, int LDB, int LDC, int FLAGS>
__global__ __launch_bounds__(256)
void gemm_wmma(const float* __restrict__ A, const float* __restrict__ A2,
               const float* __restrict__ B, const float* __restrict__ Res,
               float* __restrict__ C, int K,
               int nh, long sAb, long sAh, long sBb, long sBh,
               long sCb, long sCh)
{
    constexpr int WAVES_M = BM / WM;         // 4
    constexpr int AM_FR   = WM / 16;         // 2
    constexpr int BN_FR   = WN / 16;         // 4 (BN=128) or 2 (BN=64)
    constexpr int A_STR   = 36;              // bf16 per LDS A row (32 + 4 pad)

    __shared__ __bf16 a_lds[2][BM * A_STR];
    __shared__ __bf16 b_lds[2][32 * BN];     // pair-interleaved: uint idx = (k>>1)*BN + n

    const int tid  = threadIdx.x;
    const int lane = tid & 31;
    const int wid  = tid >> 5;
    const int wm   = wid % WAVES_M;
    const int wn   = wid / WAVES_M;

    const int bz = blockIdx.z;
    const int bb = bz / nh;
    const int hh = bz % nh;
    const float* Ap  = A + bb * sAb + hh * sAh;
    const float* A2p = (FLAGS & FLAG_GEGLU) ? (A2 + bb * sAb + hh * sAh) : nullptr;
    const float* Bp  = B + bb * sBb + hh * sBh;
    float*       Cp  = C + bb * sCb + hh * sCh;
    const float* Rp  = (FLAGS & FLAG_RES) ? (Res + bb * sCb + hh * sCh) : nullptr;

    const int rowBase = blockIdx.y * BM;
    const int colBase = blockIdx.x * BN;

    v8f acc[AM_FR][BN_FR];
#pragma unroll
    for (int i = 0; i < AM_FR; ++i)
#pragma unroll
        for (int j = 0; j < BN_FR; ++j)
            acc[i][j] = (v8f){0.f, 0.f, 0.f, 0.f, 0.f, 0.f, 0.f, 0.f};

    const int kb2a = (lane >> 4) ? 4 : 0;    // A: lanes 16..31 start at K=8  -> pair 4
    const int kb2b = (lane >> 4) ? 4 : 0;    // B: lanes 16..31 start at K=8  -> pair 4
    const int lane15 = lane & 15;

    // ---- stage one 32-wide K tile into LDS buffer `buf` (all 32-bit offsets)
    auto stage = [&](int k0, int buf) {
#pragma unroll
        for (int it = 0; it < (BM * 32) / (256 * 4); ++it) {
            const int e  = tid + it * 256;
            const int m  = e >> 3;           // 8 float4-groups per 32-wide row
            const int k4 = (e & 7) * 4;
            const int gi = (rowBase + m) * LDA + (k0 + k4);
            float4 a4 = *(const float4*)(Ap + gi);
            if (FLAGS & FLAG_GEGLU) {
                const float4 u4 = *(const float4*)(A2p + gi);
                a4.x = gelu_exact(a4.x) * u4.x;
                a4.y = gelu_exact(a4.y) * u4.y;
                a4.z = gelu_exact(a4.z) * u4.z;
                a4.w = gelu_exact(a4.w) * u4.w;
            }
            unsigned* dst = (unsigned*)(a_lds[buf] + m * A_STR + k4);
            dst[0] = pack_bf16(a4.x, a4.y);
            dst[1] = pack_bf16(a4.z, a4.w);
        }
        if (FLAGS & FLAG_BTRANS) {           // B stored [N,K]: float4 along K
            unsigned* bw = (unsigned*)b_lds[buf];
#pragma unroll
            for (int it = 0; it < (32 * BN) / (256 * 4); ++it) {
                const int e  = tid + it * 256;
                const int nn = e >> 3;
                const int k4 = (e & 7) * 4;
                const float4 b4 =
                    *(const float4*)(Bp + (colBase + nn) * LDB + (k0 + k4));
                bw[(k4 >> 1) * BN + nn]       = pack_bf16(b4.x, b4.y);
                bw[((k4 >> 1) + 1) * BN + nn] = pack_bf16(b4.z, b4.w);
            }
        } else {                             // B stored [K,N]: float4 along N
#pragma unroll
            for (int it = 0; it < (32 * BN) / (256 * 4); ++it) {
                const int e  = tid + it * 256;
                const int kk = e / (BN / 4);
                const int n4 = (e % (BN / 4)) * 4;
                const float4 b4 =
                    *(const float4*)(Bp + (k0 + kk) * LDB + (colBase + n4));
                __bf16* d = b_lds[buf] + ((kk >> 1) * BN + n4) * 2 + (kk & 1);
                d[0] = (__bf16)b4.x;
                d[2] = (__bf16)b4.y;
                d[4] = (__bf16)b4.z;
                d[6] = (__bf16)b4.w;
            }
        }
    };

    // ---- fragment loads (CDNA5 wave32 layouts) + 16x16x32 bf16 WMMAs
    auto compute = [&](int buf) {
        v16bf afrag[AM_FR];
#pragma unroll
        for (int i = 0; i < AM_FR; ++i) {
            const int m = wm * WM + i * 16 + lane15;
            const unsigned* au = (const unsigned*)(a_lds[buf] + m * A_STR);
            v8u u;
#pragma unroll
            for (int t = 0; t < 4; ++t) {
                u[t]     = au[kb2a + t];         // K pairs 0..7   (this lane half)
                u[t + 4] = au[8 + kb2a + t];     // K pairs 16..31 (this lane half)
            }
            afrag[i] = __builtin_bit_cast(v16bf, u);
        }
        v16bf bfrag[BN_FR];
        const unsigned* br = (const unsigned*)b_lds[buf];
#pragma unroll
        for (int j = 0; j < BN_FR; ++j) {
            const int n = wn * WN + j * 16 + lane15;
            v8u u;
#pragma unroll
            for (int t = 0; t < 8; ++t)
                u[t] = br[(kb2b + (t & 3) + ((t >> 2) << 3)) * BN + n];
            bfrag[j] = __builtin_bit_cast(v16bf, u);
        }
#pragma unroll
        for (int i = 0; i < AM_FR; ++i)
#pragma unroll
            for (int j = 0; j < BN_FR; ++j)
                acc[i][j] = __builtin_amdgcn_wmma_f32_16x16x32_bf16(
                    false, afrag[i], false, bfrag[j],
                    (short)0, acc[i][j], false, false);
    };

    // ---- software pipeline: stage(s+1) overlaps compute(s), 1 barrier/step
    const int steps = K >> 5;
    stage(0, 0);
    __syncthreads();
    for (int s = 0; s < steps; ++s) {
        if (s + 1 < steps) stage((s + 1) << 5, (s + 1) & 1);
        compute(s & 1);
        __syncthreads();
    }

    // ---- epilogue: C/D layout = lane gives column, VGPR r gives row ----
    const int rhalf = (lane >> 4) * 8;
#pragma unroll
    for (int i = 0; i < AM_FR; ++i)
#pragma unroll
        for (int j = 0; j < BN_FR; ++j) {
            const int r0 = rowBase + wm * WM + i * 16 + rhalf;
            const int c0 = colBase + wn * WN + j * 16 + lane15;
#pragma unroll
            for (int r = 0; r < 8; ++r) {
                const int off = (r0 + r) * LDC + c0;   // 32-bit addressing
                float v = acc[i][j][r];
                if (FLAGS & FLAG_RES) v += Rp[off];
                Cp[off] = v;
            }
        }
}

// ---------------------------------------------------------------------------
// Wave-per-token LayerNorm (768 = 32 lanes x 24).
// ---------------------------------------------------------------------------
__device__ __forceinline__ float wave_sum(float s) {
#pragma unroll
    for (int o = 16; o > 0; o >>= 1) s += __shfl_xor(s, o, 32);
    return s;
}

__global__ __launch_bounds__(256)
void ln_kernel(const float* __restrict__ in, const float* __restrict__ gamma,
               float* __restrict__ out, int ntok)
{
    const int wave = threadIdx.x >> 5, lane = threadIdx.x & 31;
    const int t = blockIdx.x * 8 + wave;
    if (t >= ntok) return;
    const float* row = in + t * HID;
    float v[24]; float s = 0.f;
#pragma unroll
    for (int i = 0; i < 24; ++i) { v[i] = row[lane + i * 32]; s += v[i]; }
    const float mean = wave_sum(s) * (1.0f / HID);
    float vs = 0.f;
#pragma unroll
    for (int i = 0; i < 24; ++i) { const float d = v[i] - mean; vs += d * d; }
    const float inv = rsqrtf(wave_sum(vs) * (1.0f / HID) + 1e-12f);
    float* orow = out + t * HID;
#pragma unroll
    for (int i = 0; i < 24; ++i) {
        const int c = lane + i * 32;
        orow[c] = (v[i] - mean) * inv * gamma[c];
    }
}

// ---------------------------------------------------------------------------
// Embedding gather + LayerNorm (wave per token).
// ---------------------------------------------------------------------------
__global__ __launch_bounds__(256)
void embed_kernel(const int* __restrict__ ids, const int* __restrict__ tt,
                  const float* __restrict__ wemb, const float* __restrict__ temb,
                  const float* __restrict__ gamma, float* __restrict__ out)
{
    const int wave = threadIdx.x >> 5, lane = threadIdx.x & 31;
    const int t = blockIdx.x * 8 + wave;
    if (t >= TOK) return;
    const float* wr = wemb + (long)ids[t] * HID;
    const float* tr = temb + (long)tt[t] * HID;
    float v[24]; float s = 0.f;
#pragma unroll
    for (int i = 0; i < 24; ++i) {
        const int c = lane + i * 32;
        v[i] = wr[c] + tr[c]; s += v[i];
    }
    const float mean = wave_sum(s) * (1.0f / HID);
    float vs = 0.f;
#pragma unroll
    for (int i = 0; i < 24; ++i) { const float d = v[i] - mean; vs += d * d; }
    const float inv = rsqrtf(wave_sum(vs) * (1.0f / HID) + 1e-12f);
    float* orow = out + t * HID;
#pragma unroll
    for (int i = 0; i < 24; ++i) {
        const int c = lane + i * 32;
        orow[c] = (v[i] - mean) * inv * gamma[c];
    }
}

// ---------------------------------------------------------------------------
// Masked softmax over keys (wave per (b,h,q) row): p = softmax(s*scale + bias).
// bias = pad (mask==0 -> -1e9) + band (|q-k|>64 -> -1e9, windowed layers only).
// ---------------------------------------------------------------------------
__global__ __launch_bounds__(256)
void softmax_kernel(float* __restrict__ sc, const int* __restrict__ mask,
                    int isGlob, float scale)
{
    const int wave = threadIdx.x >> 5, lane = threadIdx.x & 31;
    const int r = blockIdx.x * 8 + wave;           // r = (b*NH + h)*SEQ + q
    if (r >= BATCH * NH * SEQ) return;
    const int q = r % SEQ;
    const int b = r / (NH * SEQ);
    float* p = sc + (long)r * SEQ;
    float v[16]; float mx = NEGB;
#pragma unroll
    for (int i = 0; i < 16; ++i) {
        const int k = lane + i * 32;
        float bias = (mask[b * SEQ + k] == 0) ? NEGB : 0.f;
        if (!isGlob && (k - q > 64 || q - k > 64)) bias = NEGB;
        v[i] = p[k] * scale + bias;
        mx = fmaxf(mx, v[i]);
    }
#pragma unroll
    for (int o = 16; o > 0; o >>= 1) mx = fmaxf(mx, __shfl_xor(mx, o, 32));
    float s = 0.f;
#pragma unroll
    for (int i = 0; i < 16; ++i) { v[i] = __expf(v[i] - mx); s += v[i]; }
    const float inv = 1.0f / wave_sum(s);
#pragma unroll
    for (int i = 0; i < 16; ++i) p[lane + i * 32] = v[i] * inv;
}

// ---------------------------------------------------------------------------
// Host-side orchestration.
// ---------------------------------------------------------------------------
extern "C" void kernel_launch(void* const* d_in, const int* in_sizes, int n_in,
                              void* d_out, int out_size, void* d_ws, size_t ws_size,
                              hipStream_t stream)
{
    (void)in_sizes; (void)n_in; (void)out_size; (void)ws_size;

    const int*   input_ids = (const int*)  d_in[0];
    const int*   attn_mask = (const int*)  d_in[1];
    const int*   tok_type  = (const int*)  d_in[2];
    const float* word_emb  = (const float*)d_in[3];
    const float* type_emb  = (const float*)d_in[4];
    const float* emb_ln_g  = (const float*)d_in[5];
    const float* Wq        = (const float*)d_in[6];
    const float* Wk        = (const float*)d_in[7];
    const float* Wv        = (const float*)d_in[8];
    const float* Wo        = (const float*)d_in[9];
    const float* ln1_g     = (const float*)d_in[10];
    const float* ln2_g     = (const float*)d_in[11];
    const float* Wg        = (const float*)d_in[12];
    const float* Wu        = (const float*)d_in[13];
    const float* Wd        = (const float*)d_in[14];
    const float* fln_g     = (const float*)d_in[15];

    // workspace layout (floats): ~26.8M elements ~= 107 MB
    float* ws = (float*)d_ws;
    float* xb = ws;                      // [2048,768] residual stream
    float* hb = xb + (long)TOK * HID;    // [2048,768] LN output
    float* qb = hb + (long)TOK * HID;
    float* kb = qb + (long)TOK * HID;
    float* vb = kb + (long)TOK * HID;
    float* ab = vb + (long)TOK * HID;    // attention output (pre-O)
    float* gb = ab + (long)TOK * HID;    // [2048,1152]
    float* ub = gb + (long)TOK * FF;     // [2048,1152]
    float* sb = ub + (long)TOK * FF;     // scores [B,NH,S,S] = 12.58M

    const dim3 blk(256);
    const dim3 gLN(TOK / 8);                     // 256 blocks, wave/token
    const dim3 gSM(BATCH * NH * SEQ / 8);        // 3072 blocks
    const dim3 g768(HID / 128, TOK / 128, 1);    // (6,16)
    const dim3 g1152(FF / 128, TOK / 128, 1);    // (9,16)
    const dim3 gSC(SEQ / 128, SEQ / 128, BATCH * NH);  // (4,4,48)
    const dim3 gPV(DH / 64, SEQ / 128, BATCH * NH);    // (1,4,48)

    const long sQKb = (long)SEQ * HID;   // batch stride in q/k/v/attn buffers
    const long sSCb = (long)NH * SEQ * SEQ;
    const long sSCh = (long)SEQ * SEQ;
    const float scale = 0.125f;          // 1/sqrt(64)

    embed_kernel<<<gLN, blk, 0, stream>>>(input_ids, tok_type, word_emb, type_emb,
                                          emb_ln_g, xb);

    for (int l = 0; l < NL; ++l) {
        const int isGlob = ((l + 1) % 3 == 0);
        const long wOff  = (long)l * HID * HID;
        const long wOffF = (long)l * HID * FF;

        // LN1
        ln_kernel<<<gLN, blk, 0, stream>>>(xb, ln1_g + (long)l * HID, hb, TOK);

        // Q, K, V projections
        gemm_wmma<128,128,32,64, HID,HID,HID, 0><<<g768, blk, 0, stream>>>(
            hb, nullptr, Wq + wOff, nullptr, qb, HID, 1, 0,0,0,0,0,0);
        gemm_wmma<128,128,32,64, HID,HID,HID, 0><<<g768, blk, 0, stream>>>(
            hb, nullptr, Wk + wOff, nullptr, kb, HID, 1, 0,0,0,0,0,0);
        gemm_wmma<128,128,32,64, HID,HID,HID, 0><<<g768, blk, 0, stream>>>(
            hb, nullptr, Wv + wOff, nullptr, vb, HID, 1, 0,0,0,0,0,0);

        // scores[b,h] = Q[b,:,h,:] @ K[b,:,h,:]^T   (batched over 48 heads)
        gemm_wmma<128,128,32,64, HID,HID,SEQ, FLAG_BTRANS><<<gSC, blk, 0, stream>>>(
            qb, nullptr, kb, nullptr, sb, DH,
            NH, sQKb, DH, sQKb, DH, sSCb, sSCh);

        // masked softmax (in-place)
        softmax_kernel<<<gSM, blk, 0, stream>>>(sb, attn_mask, isGlob, scale);

        // attn[b,:,h,:] = P[b,h] @ V[b,:,h,:]
        gemm_wmma<128,64,32,32, SEQ,HID,HID, 0><<<gPV, blk, 0, stream>>>(
            sb, nullptr, vb, nullptr, ab, SEQ,
            NH, sSCb, sSCh, sQKb, DH, sQKb, DH);

        // x += attn @ Wo
        gemm_wmma<128,128,32,64, HID,HID,HID, FLAG_RES><<<g768, blk, 0, stream>>>(
            ab, nullptr, Wo + wOff, xb, xb, HID, 1, 0,0,0,0,0,0);

        // LN2
        ln_kernel<<<gLN, blk, 0, stream>>>(xb, ln2_g + (long)l * HID, hb, TOK);

        // gate / up projections
        gemm_wmma<128,128,32,64, HID,FF,FF, 0><<<g1152, blk, 0, stream>>>(
            hb, nullptr, Wg + wOffF, nullptr, gb, HID, 1, 0,0,0,0,0,0);
        gemm_wmma<128,128,32,64, HID,FF,FF, 0><<<g1152, blk, 0, stream>>>(
            hb, nullptr, Wu + wOffF, nullptr, ub, HID, 1, 0,0,0,0,0,0);

        // x += (gelu(g)*u) @ Wd   (GeGLU fused into A staging)
        gemm_wmma<128,128,32,64, FF,HID,HID, FLAG_GEGLU|FLAG_RES>
            <<<g768, blk, 0, stream>>>(
            gb, ub, Wd + wOffF, xb, xb, FF, 1, 0,0,0,0,0,0);
    }

    // final LN -> d_out (fp32)
    ln_kernel<<<gLN, blk, 0, stream>>>(xb, fln_g, (float*)d_out, TOK);
}
